// HRGATConv_67044439491161
// MI455X (gfx1250) — matrix operand order repaired and checked by reading, not compiled, not run on hardware
//
#include <hip/hip_runtime.h>
#include <hip/hip_bf16.h>

#define N_NODES 100000
#define IN_DIM  128
#define OUT_DIM 128
#define N_REL   8
#define N_BASIS 4
#define N_EDGES 1600000
#define LEAKY   0.2f

typedef __attribute__((ext_vector_type(16))) __bf16 v16bf;
typedef __attribute__((ext_vector_type(8)))  __bf16 v8bf;
typedef __attribute__((ext_vector_type(8)))  float  v8f;

// ---------------------------------------------------------------------------
// Kernel 1: convert node embeddings to bf16 (X is reused 8x; bf16 halves the
// GEMM read traffic and unlocks v_wmma_f32_16x16x32_bf16).
// ---------------------------------------------------------------------------
__global__ void cvt_x_bf16(const float* __restrict__ x, __bf16* __restrict__ xb, int n) {
    int i = blockIdx.x * blockDim.x + threadIdx.x;
    if (i < n) xb[i] = (__bf16)x[i];
}

// ---------------------------------------------------------------------------
// Kernel 2: weights[r] = sum_b att[r,b] * basis[b], stored TRANSPOSED
// (Wt[r][o][i]) in bf16 so the WMMA B-fragment loads are contiguous B128s.
// ---------------------------------------------------------------------------
__global__ void make_weights(const float* __restrict__ att,
                             const float* __restrict__ basis,
                             __bf16* __restrict__ wt) {
    int idx = blockIdx.x * blockDim.x + threadIdx.x;       // r*IN*OUT threads
    if (idx >= N_REL * IN_DIM * OUT_DIM) return;
    int r = idx / (IN_DIM * OUT_DIM);
    int rem = idx - r * (IN_DIM * OUT_DIM);
    int i = rem / OUT_DIM;
    int o = rem - i * OUT_DIM;
    float acc = 0.f;
#pragma unroll
    for (int b = 0; b < N_BASIS; ++b)
        acc += att[r * N_BASIS + b] * basis[((size_t)b * IN_DIM + i) * OUT_DIM + o];
    wt[((size_t)r * OUT_DIM + o) * IN_DIM + i] = (__bf16)acc;
}

// ---------------------------------------------------------------------------
// Kernel 3: Wh[r] = X(100000x128) @ W[r](128x128) via WMMA.
// grid = (6250 M-tiles, 8 relations), block = 256 (8 waves); wave w owns
// column tile w (16 cols), so each block emits a full 16x128 strip.
// Fragment layouts follow ISA 7.12.2 (wave32):
//   A 16x32 bf16: lane L row M=L%16; half h -> K=(h&7)+16*(h>>3)+8*(L>=16)
//   B 32x16 bf16: lane L col N=L%16; half h -> K=(h&7)+16*(h>>3)+8*(L>=16)
//   C/D f32:      vgpr v -> M=v+8*(L>=16), N=L%16
// ---------------------------------------------------------------------------
__global__ void wmma_gemm(const __bf16* __restrict__ xb,
                          const __bf16* __restrict__ wt,
                          float* __restrict__ wh) {
    const int lane = threadIdx.x & 31;
    const int wave = threadIdx.x >> 5;
    const int hi   = lane >> 4;        // 0: lanes 0-15, 1: lanes 16-31
    const int m    = lane & 15;
    const int row0 = blockIdx.x * 16;
    const int r    = blockIdx.y;
    const int col0 = wave * 16;

    const __bf16* xrow = xb + (size_t)(row0 + m) * IN_DIM;                 // A row (M)
    const __bf16* wrow = wt + ((size_t)r * OUT_DIM + (col0 + m)) * IN_DIM; // B col (N)

    v8f c = {};
#pragma unroll
    for (int k0 = 0; k0 < IN_DIM; k0 += 32) {
        v8bf a_lo = *(const v8bf*)(xrow + k0 + 8 * hi);        // h=0..7 : K=k0+8*hi..+7
        v8bf a_hi = *(const v8bf*)(xrow + k0 + 16 + 8 * hi);   // h=8..15: K=k0+16+8*hi..+7
        v8bf b_lo = *(const v8bf*)(wrow + k0 + 8 * hi);        // h=0..7 : K=k0+8*hi..+7
        v8bf b_hi = *(const v8bf*)(wrow + k0 + 16 + 8 * hi);   // h=8..15: K=k0+16+8*hi..+7
        v16bf a, b;
#pragma unroll
        for (int i = 0; i < 8; ++i) {
            a[i] = a_lo[i]; a[i + 8] = a_hi[i];
            b[i] = b_lo[i]; b[i + 8] = b_hi[i];
        }
        c = __builtin_amdgcn_wmma_f32_16x16x32_bf16(
                /*neg_a=*/false, a, /*neg_b=*/false, b,
                /*c_mod=*/(short)0, c, /*reuse_a=*/false, /*reuse_b=*/false);
    }

    float* orow = wh + ((size_t)r * N_NODES + row0) * OUT_DIM + col0 + m;
#pragma unroll
    for (int v = 0; v < 8; ++v)
        orow[(size_t)(v + 8 * hi) * OUT_DIM] = c[v];
}

// ---------------------------------------------------------------------------
// Kernel 4: s_src[r,n] = <Wh[r,n,:], a_src[r,:]>, s_dst likewise.
// One wave per (r,n); lane handles 4 consecutive floats; shfl_xor reduce.
// ---------------------------------------------------------------------------
__global__ void score_kernel(const float* __restrict__ wh,
                             const float* __restrict__ attention,
                             float* __restrict__ s_src,
                             float* __restrict__ s_dst) {
    int gw   = (blockIdx.x * blockDim.x + threadIdx.x) >> 5;   // wave id = (r,n)
    int lane = threadIdx.x & 31;
    if (gw >= N_REL * N_NODES) return;
    int r = gw / N_NODES;
    const float* row  = wh + (size_t)gw * OUT_DIM + lane * 4;
    const float* asrc = attention + (size_t)r * 2 * OUT_DIM + lane * 4;
    const float* adst = asrc + OUT_DIM;
    float4 w  = *(const float4*)row;
    float4 as = *(const float4*)asrc;
    float4 ad = *(const float4*)adst;
    float ss = w.x * as.x + w.y * as.y + w.z * as.z + w.w * as.w;
    float sd = w.x * ad.x + w.y * ad.y + w.z * ad.z + w.w * ad.w;
#pragma unroll
    for (int off = 16; off > 0; off >>= 1) {
        ss += __shfl_xor(ss, off, 32);
        sd += __shfl_xor(sd, off, 32);
    }
    if (lane == 0) { s_src[gw] = ss; s_dst[gw] = sd; }
}

// ---------------------------------------------------------------------------
// Kernel 5a: init segment buffers (e_max = -inf, denom = 0).
// ---------------------------------------------------------------------------
__global__ void init_seg(float* __restrict__ e_max, float* __restrict__ denom, int n) {
    int i = blockIdx.x * blockDim.x + threadIdx.x;
    if (i < n) { e_max[i] = -__builtin_inff(); denom[i] = 0.f; }
}

// float atomic-max via signed/unsigned int ordering trick
__device__ __forceinline__ void atomicMaxF(float* addr, float v) {
    if (v >= 0.f) atomicMax((int*)addr, __float_as_int(v));
    else          atomicMin((unsigned int*)addr, __float_as_uint(v));
}

// ---------------------------------------------------------------------------
// Kernel 5b: e_raw = leaky_relu(s_src[rel,src] + s_dst[rel,dst]); seg-max.
// ---------------------------------------------------------------------------
__global__ void edge_raw(const int* __restrict__ triples,
                         const float* __restrict__ s_src,
                         const float* __restrict__ s_dst,
                         float* __restrict__ e_raw,
                         float* __restrict__ e_max) {
    int e = blockIdx.x * blockDim.x + threadIdx.x;
    if (e >= N_EDGES) return;
    int src = triples[3 * e + 0];
    int rel = triples[3 * e + 1];
    int dst = triples[3 * e + 2];
    float s  = s_src[(size_t)rel * N_NODES + src] + s_dst[(size_t)rel * N_NODES + dst];
    float er = s > 0.f ? s : LEAKY * s;
    e_raw[e] = er;
    atomicMaxF(e_max + (size_t)rel * N_NODES + dst, er);
}

// ---------------------------------------------------------------------------
// Kernel 5c: e_exp = exp(e_raw - e_max[seg]); denom seg-sum.
// ---------------------------------------------------------------------------
__global__ void edge_exp(const int* __restrict__ triples,
                         const float* __restrict__ e_raw,
                         const float* __restrict__ e_max,
                         float* __restrict__ e_exp,
                         float* __restrict__ denom) {
    int e = blockIdx.x * blockDim.x + threadIdx.x;
    if (e >= N_EDGES) return;
    int rel = triples[3 * e + 1];
    int dst = triples[3 * e + 2];
    size_t seg = (size_t)rel * N_NODES + dst;
    float ex = __expf(e_raw[e] - e_max[seg]);
    e_exp[e] = ex;
    atomicAdd(denom + seg, ex);
}

// ---------------------------------------------------------------------------
// Kernel 6: out[n,:] = bias + residual (pre-accumulation init).
// ---------------------------------------------------------------------------
__global__ void init_out(const float* __restrict__ x,
                         const float* __restrict__ bias,
                         float* __restrict__ out) {
    int i = blockIdx.x * blockDim.x + threadIdx.x;
    if (i < N_NODES * OUT_DIM) out[i] = bias[i & (OUT_DIM - 1)] + x[i];
}

// ---------------------------------------------------------------------------
// Kernel 7: scatter messages: out[dst,:] += alpha_e * Wh[rel,src,:].
// One wave per edge; lane owns 4 consecutive floats (float4 gather + atomics).
// ---------------------------------------------------------------------------
__global__ void scatter_msgs(const int* __restrict__ triples,
                             const float* __restrict__ wh,
                             const float* __restrict__ e_exp,
                             const float* __restrict__ denom,
                             float* __restrict__ out) {
    int gw   = (blockIdx.x * blockDim.x + threadIdx.x) >> 5;   // edge id
    int lane = threadIdx.x & 31;
    if (gw >= N_EDGES) return;
    int src = triples[3 * gw + 0];
    int rel = triples[3 * gw + 1];
    int dst = triples[3 * gw + 2];
    size_t seg = (size_t)rel * N_NODES + dst;
    float alpha = e_exp[gw] / fmaxf(denom[seg], 1e-8f);
    float4 v = *(const float4*)(wh + ((size_t)rel * N_NODES + src) * OUT_DIM + lane * 4);
    float* o = out + (size_t)dst * OUT_DIM + lane * 4;
    atomicAdd(o + 0, alpha * v.x);
    atomicAdd(o + 1, alpha * v.y);
    atomicAdd(o + 2, alpha * v.z);
    atomicAdd(o + 3, alpha * v.w);
}

// ---------------------------------------------------------------------------
extern "C" void kernel_launch(void* const* d_in, const int* in_sizes, int n_in,
                              void* d_out, int out_size, void* d_ws, size_t ws_size,
                              hipStream_t stream) {
    const float* x         = (const float*)d_in[0];   // node_embeddings (N, IN)
    const int*   triples   = (const int*)  d_in[1];   // (E, 3)
    const float* basis     = (const float*)d_in[3];   // (B, IN, OUT)
    const float* att       = (const float*)d_in[4];   // (R, B)
    const float* attention = (const float*)d_in[5];   // (R, 2*OUT)
    const float* bias      = (const float*)d_in[6];   // (OUT,)
    float* out = (float*)d_out;

    // workspace carve-up (all sizes 256B-aligned)
    char* ws = (char*)d_ws;
    __bf16* xb    = (__bf16*)ws;                          ws += (size_t)N_NODES * IN_DIM * 2;          // 25.6 MB
    __bf16* wt    = (__bf16*)ws;                          ws += (size_t)N_REL * IN_DIM * OUT_DIM * 2;  // 256 KB
    float*  wh    = (float*)ws;                           ws += (size_t)N_REL * N_NODES * OUT_DIM * 4; // 409.6 MB
    float*  s_src = (float*)ws;                           ws += (size_t)N_REL * N_NODES * 4;           // 3.2 MB
    float*  s_dst = (float*)ws;                           ws += (size_t)N_REL * N_NODES * 4;
    float*  e_max = (float*)ws;                           ws += (size_t)N_REL * N_NODES * 4;
    float*  denom = (float*)ws;                           ws += (size_t)N_REL * N_NODES * 4;
    float*  e_raw = (float*)ws;                           ws += (size_t)N_EDGES * 4;                   // 6.4 MB
    float*  e_exp = (float*)ws;                           ws += (size_t)N_EDGES * 4;

    const int T = 256;

    // precision conversion + basis combine
    cvt_x_bf16  <<<(N_NODES * IN_DIM + T - 1) / T, T, 0, stream>>>(x, xb, N_NODES * IN_DIM);
    make_weights<<<(N_REL * IN_DIM * OUT_DIM + T - 1) / T, T, 0, stream>>>(att, basis, wt);

    // Wh = X @ W[r] via bf16 WMMA, f32 accumulate
    dim3 ggrid(N_NODES / 16, N_REL);   // 6250 x 8
    wmma_gemm<<<ggrid, T, 0, stream>>>(xb, wt, wh);

    // attention scores per (r, n)
    score_kernel<<<(N_REL * N_NODES) / (T / 32), T, 0, stream>>>(wh, attention, s_src, s_dst);

    // segment softmax over edges
    init_seg<<<(N_REL * N_NODES + T - 1) / T, T, 0, stream>>>(e_max, denom, N_REL * N_NODES);
    edge_raw<<<(N_EDGES + T - 1) / T, T, 0, stream>>>(triples, s_src, s_dst, e_raw, e_max);
    edge_exp<<<(N_EDGES + T - 1) / T, T, 0, stream>>>(triples, e_raw, e_max, e_exp, denom);

    // out = bias + residual, then scatter alpha * Wh[rel,src]
    init_out<<<(N_NODES * OUT_DIM + T - 1) / T, T, 0, stream>>>(x, bias, out);
    scatter_msgs<<<(N_EDGES * 32 + T - 1) / T, T, 0, stream>>>(triples, wh, e_exp, denom, out);
}
